// DRAW_21784074125631
// MI455X (gfx1250) — matrix operand order repaired and checked by reading, not compiled, not run on hardware
//
#include <hip/hip_runtime.h>
#include <hip/hip_bf16.h>

// DRAW (sequential VAE) for MI455X / gfx1250.
// ~335 GFLOP/call over 32 sequential steps; weights ~40MB bf16 (L2-resident).
// Matrix-core bound -> bf16 WMMA 16x16x32 f32-acc, branch-free GEMM waves.

#define BSZ    256
#define IMGD   64
#define RSZ    12
#define HID    1024
#define LATD   256
#define TSTEPS 32
#define ENC_IN 2336          // 2*12*12 + 2*1024
#define GATE   4096          // 4*HID
#define WN_PAD 192           // writer N (144) padded to multiple of 64

typedef __attribute__((ext_vector_type(16))) __bf16 v16bf;
typedef __attribute__((ext_vector_type(8)))  __bf16 v8bf;
typedef __attribute__((ext_vector_type(8)))  float  v8f;

union AFrag { v16bf v; v8bf h[2]; };

__device__ __forceinline__ float sigf(float x) { return 1.0f / (1.0f + __expf(-x)); }
__device__ __forceinline__ __bf16 f2bf(float x) { return (__bf16)x; }

// ---------------------------------------------------------------- convert
__global__ void k_f32_to_bf16(const float* __restrict__ src, __bf16* __restrict__ dst, int n) {
    int i = blockIdx.x * blockDim.x + threadIdx.x;
    if (i < n) dst[i] = f2bf(src[i]);
}

// pad bias to np entries (zeros beyond n)
__global__ void k_pad_bias(const float* __restrict__ src, float* __restrict__ dst, int n, int np) {
    int i = blockIdx.x * blockDim.x + threadIdx.x;
    if (i < np) dst[i] = (i < n) ? src[i] : 0.0f;
}

// ---------------------------------------------------------------- WMMA GEMM
// C[M,N] = bias[N] + A1[M,K1] @ W1[N,K1]^T (+ A2[M,K2] @ W2[N,K2]^T)
// Requirements (guaranteed by host): N % 64 == 0, K1/K2 % 32 == 0,
// grid covers exactly (M/16)*(N/64) waves, 8 waves per block.
// Branch-free except uniform scalar K loops -> EXEC stays all-ones for WMMA.
__global__ void __launch_bounds__(256) k_wmma_gemm_dual(
        const __bf16* __restrict__ A1, int K1,
        const __bf16* __restrict__ A2, int K2,
        const __bf16* __restrict__ W1,
        const __bf16* __restrict__ W2,
        const float* __restrict__ bias,
        float* __restrict__ C, int N, int ngroups) {
    int lane = threadIdx.x & 31;
    // wave id forced into an SGPR so mt/ng and base addresses are scalar
    int wid = __builtin_amdgcn_readfirstlane((int)(blockIdx.x * 8) + (threadIdx.x >> 5));
    int mt = wid / ngroups;            // M tile (16 rows)
    int ng = wid - mt * ngroups;       // group of 4 N-tiles
    int n0 = ng * 64;

    int col = lane & 15;
    int hi  = lane >> 4;               // half-wave select
    int row = mt * 16 + col;           // A row for this lane

    v8f acc[4];
#pragma unroll
    for (int j = 0; j < 4; ++j) {
        float bv = bias[n0 + j * 16 + col];
#pragma unroll
        for (int r = 0; r < 8; ++r) acc[j][r] = bv;
    }

    // ---- pass 1: A1 @ W1^T
    {
        const __bf16* Arow = A1 + (size_t)row * K1 + hi * 16;
        const __bf16* Wb   = W1 + (size_t)(n0 + col) * K1 + hi * 16;
        const size_t  wstr = (size_t)16 * K1;      // stride between N-tiles
        for (int k = 0; k < K1; k += 32) {
            AFrag a;
            a.h[0] = *(const v8bf*)(Arow + k);         // K k..k+7  / k+16..k+23
            a.h[1] = *(const v8bf*)(Arow + k + 8);     // K k+8..   / k+24..
#pragma unroll
            for (int j = 0; j < 4; ++j) {
                v16bf b = *(const v16bf*)(Wb + (size_t)j * wstr + k);
                acc[j] = __builtin_amdgcn_wmma_f32_16x16x32_bf16(
                             false, a.v, false, b, (short)0, acc[j], false, false);
            }
        }
    }
    // ---- pass 2: A2 @ W2^T (hidden-state term of LSTM); K2 uniform scalar
    if (K2 > 0) {
        const __bf16* Arow = A2 + (size_t)row * K2 + hi * 16;
        const __bf16* Wb   = W2 + (size_t)(n0 + col) * K2 + hi * 16;
        const size_t  wstr = (size_t)16 * K2;
        for (int k = 0; k < K2; k += 32) {
            AFrag a;
            a.h[0] = *(const v8bf*)(Arow + k);
            a.h[1] = *(const v8bf*)(Arow + k + 8);
#pragma unroll
            for (int j = 0; j < 4; ++j) {
                v16bf b = *(const v16bf*)(Wb + (size_t)j * wstr + k);
                acc[j] = __builtin_amdgcn_wmma_f32_16x16x32_bf16(
                             false, a.v, false, b, (short)0, acc[j], false, false);
            }
        }
    }
    // ---- store: C/D layout VGPR r -> M = r + 8*hi, N = lane&15
#pragma unroll
    for (int j = 0; j < 4; ++j) {
        float* Cp = C + (size_t)(mt * 16 + hi * 8) * N + n0 + j * 16 + col;
#pragma unroll
        for (int r = 0; r < 8; ++r) Cp[(size_t)r * N] = acc[j][r];
    }
}

// ---------------------------------------------------------------- attention proj
// res[B,5] = hcat[B,2048] @ Wa[5,2048]^T + ba.  One block/batch, 5 waves.
__global__ void k_attn_proj(const float* __restrict__ hcat,
                            const float* __restrict__ Wa,
                            const float* __restrict__ ba,
                            float* __restrict__ res) {
    int b    = blockIdx.x;
    int wave = threadIdx.x >> 5;   // 0..4 -> output row
    int lane = threadIdx.x & 31;
    const float* h = hcat + (size_t)b * (2 * HID);
    const float* w = Wa + (size_t)wave * (2 * HID);
    float s = 0.0f;
    for (int k = lane; k < 2 * HID; k += 32) s += h[k] * w[k];
#pragma unroll
    for (int o = 16; o > 0; o >>= 1) s += __shfl_xor(s, o, 32);
    if (lane == 0) res[b * 5 + wave] = s + ba[wave];
}

// ---------------------------------------------------------------- filter bank
// Fx/Fy: (B, 12, 64).   Fx = exp(-(g-mu_x)^2/2 * var)
__global__ void k_filterbank(const float* __restrict__ res,
                             float* __restrict__ Fx, float* __restrict__ Fy) {
    int b = blockIdx.x;
    float gx_ = res[b * 5 + 0], gy_ = res[b * 5 + 1];
    float lv  = res[b * 5 + 2], ld  = res[b * 5 + 3];
    float gx = (IMGD + 1) * (gx_ + 1.0f) * 0.5f;
    float gy = (IMGD + 1) * (gy_ + 1.0f) * 0.5f;
    float stride = (IMGD - 1) * __expf(ld) / (float)(RSZ - 1);
    float var = __expf(lv);
    for (int idx = threadIdx.x; idx < RSZ * IMGD; idx += blockDim.x) {
        int s = idx / IMGD, g = idx % IMGD;
        float offs = (float)s + 0.5f - (float)RSZ * 0.5f;
        float mx = gx + offs * stride;
        float my = gy + offs * stride;
        float dx = (float)g - mx;
        float dy = (float)g - my;
        Fx[(size_t)b * RSZ * IMGD + idx] = __expf(-dx * dx * 0.5f * var);
        Fy[(size_t)b * RSZ * IMGD + idx] = __expf(-dy * dy * 0.5f * var);
    }
}

// ---------------------------------------------------------------- read einsum
// r1 = Fx @ x @ Fy^T ; r2 = Fx @ (x - sig(canvas)) @ Fy^T ; scaled by exp(log_int)
__global__ void k_read(const float* __restrict__ x, const float* __restrict__ canvas,
                       const float* __restrict__ Fx, const float* __restrict__ Fy,
                       const float* __restrict__ res, __bf16* __restrict__ enc_in) {
    __shared__ float xs[IMGD * IMGD];
    __shared__ float es[IMGD * IMGD];
    __shared__ float fxs[RSZ * IMGD];
    __shared__ float fys[RSZ * IMGD];
    __shared__ float t1[RSZ * IMGD];
    __shared__ float t2[RSZ * IMGD];
    int b = blockIdx.x;
    for (int idx = threadIdx.x; idx < IMGD * IMGD; idx += blockDim.x) {
        float xv = x[(size_t)b * IMGD * IMGD + idx];
        float cv = canvas[(size_t)b * IMGD * IMGD + idx];
        xs[idx] = xv;
        es[idx] = xv - sigf(cv);
    }
    for (int idx = threadIdx.x; idx < RSZ * IMGD; idx += blockDim.x) {
        fxs[idx] = Fx[(size_t)b * RSZ * IMGD + idx];
        fys[idx] = Fy[(size_t)b * RSZ * IMGD + idx];
    }
    __syncthreads();
    for (int idx = threadIdx.x; idx < RSZ * IMGD; idx += blockDim.x) {
        int s = idx / IMGD, w = idx % IMGD;
        float a1 = 0.0f, a2 = 0.0f;
        for (int h = 0; h < IMGD; ++h) {
            float f = fxs[s * IMGD + h];
            a1 += f * xs[h * IMGD + w];
            a2 += f * es[h * IMGD + w];
        }
        t1[idx] = a1; t2[idx] = a2;
    }
    __syncthreads();
    float inten = __expf(res[b * 5 + 4]);
    for (int idx = threadIdx.x; idx < RSZ * RSZ; idx += blockDim.x) {
        int s = idx / RSZ, t = idx % RSZ;
        float r1 = 0.0f, r2 = 0.0f;
        for (int w = 0; w < IMGD; ++w) {
            float fy = fys[t * IMGD + w];
            r1 += t1[s * IMGD + w] * fy;
            r2 += t2[s * IMGD + w] * fy;
        }
        enc_in[(size_t)b * ENC_IN + idx]             = f2bf(inten * r1);
        enc_in[(size_t)b * ENC_IN + RSZ * RSZ + idx] = f2bf(inten * r2);
    }
}

// ---------------------------------------------------------------- LSTM pointwise
__global__ void k_lstm_pw(const float* __restrict__ gates,
                          float* __restrict__ cbuf,
                          __bf16* __restrict__ h_bf,        // [B,HID]
                          __bf16* __restrict__ hcat_bf,     // [B,2H] or null
                          float* __restrict__ hcat_f32,     // [B,2H] or null
                          __bf16* __restrict__ encin_hc) {  // enc_in base or null
    int i = blockIdx.x * blockDim.x + threadIdx.x;
    if (i >= BSZ * HID) return;
    int b = i / HID, j = i % HID;
    const float* g = gates + (size_t)b * GATE;
    float ig = g[j], fg = g[HID + j], gg = g[2 * HID + j], og = g[3 * HID + j];
    float c  = cbuf[i];
    float c2 = sigf(fg) * c + sigf(ig) * tanhf(gg);
    float h2 = sigf(og) * tanhf(c2);
    cbuf[i]  = c2;
    h_bf[i]  = f2bf(h2);
    if (hcat_bf) {
        hcat_bf[(size_t)b * 2 * HID + j]       = f2bf(h2);
        hcat_bf[(size_t)b * 2 * HID + HID + j] = f2bf(c2);
    }
    if (hcat_f32) {
        hcat_f32[(size_t)b * 2 * HID + j]       = h2;
        hcat_f32[(size_t)b * 2 * HID + HID + j] = c2;
    }
    if (encin_hc) {  // next-step encoder input cols [288, 2336)
        encin_hc[(size_t)b * ENC_IN + 2 * RSZ * RSZ + j]       = f2bf(h2);
        encin_hc[(size_t)b * ENC_IN + 2 * RSZ * RSZ + HID + j] = f2bf(c2);
    }
}

// ---------------------------------------------------------------- reparameterize
__global__ void k_z(const float* __restrict__ sq, const float* __restrict__ eps_t,
                    __bf16* __restrict__ z_bf,
                    float* __restrict__ out_mu, float* __restrict__ out_lv, int t) {
    int i = blockIdx.x * blockDim.x + threadIdx.x;
    if (i >= BSZ * LATD) return;
    int b = i / LATD, l = i % LATD;
    float mu = sq[(size_t)b * 2 * LATD + l];
    float lv = sq[(size_t)b * 2 * LATD + LATD + l];
    float z  = mu + __expf(0.5f * lv) * eps_t[(size_t)b * LATD + l];
    z_bf[i] = f2bf(z);
    out_mu[((size_t)b * TSTEPS + t) * LATD + l] = mu;
    out_lv[((size_t)b * TSTEPS + t) * LATD + l] = lv;
}

// ---------------------------------------------------------------- write einsum
// canvas += (Fy^T @ w @ Fx) / exp(log_int);  wmat has row stride WN_PAD
__global__ void k_write(const float* __restrict__ wmat, const float* __restrict__ Fx,
                        const float* __restrict__ Fy, const float* __restrict__ res,
                        float* __restrict__ canvas) {
    __shared__ float fxs[RSZ * IMGD];
    __shared__ float fys[RSZ * IMGD];
    __shared__ float wm[RSZ * RSZ];
    __shared__ float tm[RSZ * IMGD];
    int b = blockIdx.x;
    for (int idx = threadIdx.x; idx < RSZ * IMGD; idx += blockDim.x) {
        fxs[idx] = Fx[(size_t)b * RSZ * IMGD + idx];
        fys[idx] = Fy[(size_t)b * RSZ * IMGD + idx];
    }
    for (int idx = threadIdx.x; idx < RSZ * RSZ; idx += blockDim.x)
        wm[idx] = wmat[(size_t)b * WN_PAD + idx];
    __syncthreads();
    for (int idx = threadIdx.x; idx < RSZ * IMGD; idx += blockDim.x) {
        int s = idx / IMGD, w = idx % IMGD;
        float a = 0.0f;
        for (int t = 0; t < RSZ; ++t) a += wm[s * RSZ + t] * fxs[t * IMGD + w];
        tm[idx] = a;
    }
    __syncthreads();
    float inv_int = __expf(-res[b * 5 + 4]);
    for (int idx = threadIdx.x; idx < IMGD * IMGD; idx += blockDim.x) {
        int h = idx / IMGD, w = idx % IMGD;
        float a = 0.0f;
        for (int s = 0; s < RSZ; ++s) a += fys[s * IMGD + h] * tm[s * IMGD + w];
        canvas[(size_t)b * IMGD * IMGD + idx] += a * inv_int;
    }
}

// ---------------------------------------------------------------- final sigmoid
__global__ void k_sigmoid_out(const float* __restrict__ canvas, float* __restrict__ out, int n) {
    int i = blockIdx.x * blockDim.x + threadIdx.x;
    if (i < n) out[i] = sigf(canvas[i]);
}

// ================================================================ host
extern "C" void kernel_launch(void* const* d_in, const int* in_sizes, int n_in,
                              void* d_out, int out_size, void* d_ws, size_t ws_size,
                              hipStream_t stream) {
    const float* x        = (const float*)d_in[0];
    const float* eps      = (const float*)d_in[1];
    const float* enc_Wih  = (const float*)d_in[3];
    const float* enc_Whh  = (const float*)d_in[4];
    const float* enc_b    = (const float*)d_in[5];
    const float* dec_Wih  = (const float*)d_in[6];
    const float* dec_Whh  = (const float*)d_in[7];
    const float* dec_b    = (const float*)d_in[8];
    const float* q_W      = (const float*)d_in[9];
    const float* q_b      = (const float*)d_in[10];
    const float* rattn_W  = (const float*)d_in[11];
    const float* rattn_b  = (const float*)d_in[12];
    const float* wattn_W  = (const float*)d_in[13];
    const float* wattn_b  = (const float*)d_in[14];
    const float* writer_W = (const float*)d_in[15];
    const float* writer_b = (const float*)d_in[16];

    size_t off = 0;
    auto alloc = [&](size_t bytes) -> char* {
        char* p = (char*)d_ws + off;
        off += (bytes + 255) & ~(size_t)255;
        return p;
    };
    // bf16 weights (writer padded to WN_PAD rows)
    __bf16* encWih_bf  = (__bf16*)alloc(sizeof(__bf16) * (size_t)GATE * ENC_IN);
    __bf16* encWhh_bf  = (__bf16*)alloc(sizeof(__bf16) * (size_t)GATE * HID);
    __bf16* decWih_bf  = (__bf16*)alloc(sizeof(__bf16) * (size_t)GATE * LATD);
    __bf16* decWhh_bf  = (__bf16*)alloc(sizeof(__bf16) * (size_t)GATE * HID);
    __bf16* qW_bf      = (__bf16*)alloc(sizeof(__bf16) * (size_t)2 * LATD * 2 * HID);
    __bf16* writerW_bf = (__bf16*)alloc(sizeof(__bf16) * (size_t)WN_PAD * 2 * HID);
    float*  writerB_pad= (float*)alloc(sizeof(float) * WN_PAD);
    // bf16 activations
    __bf16* enc_in_bf   = (__bf16*)alloc(sizeof(__bf16) * (size_t)BSZ * ENC_IN);
    __bf16* h_enc_bf    = (__bf16*)alloc(sizeof(__bf16) * (size_t)BSZ * HID);
    __bf16* h_dec_bf    = (__bf16*)alloc(sizeof(__bf16) * (size_t)BSZ * HID);
    __bf16* hcat_enc_bf = (__bf16*)alloc(sizeof(__bf16) * (size_t)BSZ * 2 * HID);
    __bf16* hcat_dec_bf = (__bf16*)alloc(sizeof(__bf16) * (size_t)BSZ * 2 * HID);
    __bf16* z_bf        = (__bf16*)alloc(sizeof(__bf16) * (size_t)BSZ * LATD);
    // f32 state + scratch
    float* hcat_dec_f32 = (float*)alloc(sizeof(float) * (size_t)BSZ * 2 * HID);
    float* c_enc        = (float*)alloc(sizeof(float) * (size_t)BSZ * HID);
    float* c_dec        = (float*)alloc(sizeof(float) * (size_t)BSZ * HID);
    float* canvas       = (float*)alloc(sizeof(float) * (size_t)BSZ * IMGD * IMGD);
    float* gates_enc    = (float*)alloc(sizeof(float) * (size_t)BSZ * GATE);
    float* gates_dec    = (float*)alloc(sizeof(float) * (size_t)BSZ * GATE);
    float* sq           = (float*)alloc(sizeof(float) * (size_t)BSZ * 2 * LATD);
    float* wmat         = (float*)alloc(sizeof(float) * (size_t)BSZ * WN_PAD);
    float* res_r        = (float*)alloc(sizeof(float) * (size_t)BSZ * 5);
    float* res_w        = (float*)alloc(sizeof(float) * (size_t)BSZ * 5);
    float* Fx_r         = (float*)alloc(sizeof(float) * (size_t)BSZ * RSZ * IMGD);
    float* Fy_r         = (float*)alloc(sizeof(float) * (size_t)BSZ * RSZ * IMGD);
    float* Fx_w         = (float*)alloc(sizeof(float) * (size_t)BSZ * RSZ * IMGD);
    float* Fy_w         = (float*)alloc(sizeof(float) * (size_t)BSZ * RSZ * IMGD);

    float* out_canvas = (float*)d_out;
    float* out_mu = out_canvas + (size_t)BSZ * IMGD * IMGD;
    float* out_lv = out_mu + (size_t)BSZ * TSTEPS * LATD;

    // ---- weight conversion (deterministic, every call)
    auto cvt = [&](const float* s, __bf16* d, size_t n) {
        k_f32_to_bf16<<<(n + 255) / 256, 256, 0, stream>>>(s, d, (int)n);
    };
    // zero writer pad rows first, then convert the real 144 rows over them
    hipMemsetAsync(writerW_bf, 0, sizeof(__bf16) * (size_t)WN_PAD * 2 * HID, stream);
    cvt(enc_Wih,  encWih_bf,  (size_t)GATE * ENC_IN);
    cvt(enc_Whh,  encWhh_bf,  (size_t)GATE * HID);
    cvt(dec_Wih,  decWih_bf,  (size_t)GATE * LATD);
    cvt(dec_Whh,  decWhh_bf,  (size_t)GATE * HID);
    cvt(q_W,      qW_bf,      (size_t)2 * LATD * 2 * HID);
    cvt(writer_W, writerW_bf, (size_t)RSZ * RSZ * 2 * HID);
    k_pad_bias<<<1, WN_PAD, 0, stream>>>(writer_b, writerB_pad, RSZ * RSZ, WN_PAD);

    // ---- zero-init recurrent state
    hipMemsetAsync(canvas,       0, sizeof(float) * (size_t)BSZ * IMGD * IMGD, stream);
    hipMemsetAsync(c_enc,        0, sizeof(float) * (size_t)BSZ * HID, stream);
    hipMemsetAsync(c_dec,        0, sizeof(float) * (size_t)BSZ * HID, stream);
    hipMemsetAsync(hcat_dec_f32, 0, sizeof(float) * (size_t)BSZ * 2 * HID, stream);
    hipMemsetAsync(enc_in_bf,    0, sizeof(__bf16) * (size_t)BSZ * ENC_IN, stream);
    hipMemsetAsync(h_enc_bf,     0, sizeof(__bf16) * (size_t)BSZ * HID, stream);
    hipMemsetAsync(h_dec_bf,     0, sizeof(__bf16) * (size_t)BSZ * HID, stream);

    // GEMM launch: wave = 16x64 tile, 8 waves/block; grid covers waves exactly
    auto gemm = [&](const __bf16* A1, int K1, const __bf16* A2, int K2,
                    const __bf16* W1, const __bf16* W2, const float* bias,
                    float* C, int N) {
        int ngroups = N / 64;                       // 64 / 8 / 3
        int waves   = (BSZ / 16) * ngroups;         // 1024 / 128 / 48 (all %8==0)
        k_wmma_gemm_dual<<<waves / 8, 256, 0, stream>>>(A1, K1, A2, K2, W1, W2,
                                                        bias, C, N, ngroups);
    };

    for (int t = 0; t < TSTEPS; ++t) {
        // read attention (uses previous h_dec/c_dec)
        k_attn_proj<<<BSZ, 160, 0, stream>>>(hcat_dec_f32, rattn_W, rattn_b, res_r);
        k_filterbank<<<BSZ, 256, 0, stream>>>(res_r, Fx_r, Fy_r);
        k_read<<<BSZ, 256, 0, stream>>>(x, canvas, Fx_r, Fy_r, res_r, enc_in_bf);

        // encoder LSTM
        gemm(enc_in_bf, ENC_IN, h_enc_bf, HID, encWih_bf, encWhh_bf, enc_b,
             gates_enc, GATE);
        k_lstm_pw<<<(BSZ * HID + 255) / 256, 256, 0, stream>>>(
            gates_enc, c_enc, h_enc_bf, hcat_enc_bf, (float*)nullptr, (__bf16*)nullptr);

        // q projection + reparameterize
        gemm(hcat_enc_bf, 2 * HID, (const __bf16*)nullptr, 0, qW_bf,
             (const __bf16*)nullptr, q_b, sq, 2 * LATD);
        k_z<<<(BSZ * LATD + 255) / 256, 256, 0, stream>>>(
            sq, eps + (size_t)t * BSZ * LATD, z_bf, out_mu, out_lv, t);

        // decoder LSTM
        gemm(z_bf, LATD, h_dec_bf, HID, decWih_bf, decWhh_bf, dec_b, gates_dec, GATE);
        k_lstm_pw<<<(BSZ * HID + 255) / 256, 256, 0, stream>>>(
            gates_dec, c_dec, h_dec_bf, hcat_dec_bf, hcat_dec_f32, enc_in_bf);

        // writer + write attention + canvas update
        gemm(hcat_dec_bf, 2 * HID, (const __bf16*)nullptr, 0, writerW_bf,
             (const __bf16*)nullptr, writerB_pad, wmat, WN_PAD);
        k_attn_proj<<<BSZ, 160, 0, stream>>>(hcat_dec_f32, wattn_W, wattn_b, res_w);
        k_filterbank<<<BSZ, 256, 0, stream>>>(res_w, Fx_w, Fy_w);
        k_write<<<BSZ, 256, 0, stream>>>(wmat, Fx_w, Fy_w, res_w, canvas);
    }

    int npix = BSZ * IMGD * IMGD;
    k_sigmoid_out<<<(npix + 255) / 256, 256, 0, stream>>>(canvas, out_canvas, npix);
}
